// model_63101659513467
// MI455X (gfx1250) — compile-verified, matrix operand order checked by reading, not run on hardware
//
#include <hip/hip_runtime.h>

typedef __attribute__((ext_vector_type(8)))  float    v8f;
typedef __attribute__((ext_vector_type(8)))  _Float16 v8h;
typedef __attribute__((ext_vector_type(16))) _Float16 v16h;

union Frag16 {
    v16h v;
    struct { v8h lo; v8h hi; } h;
};

#define NNODES 50000
#define NEDGES 800000
#define NGRAPH 256

__device__ inline v8h zero8h() { v8h z = {}; return z; }
__device__ inline v8f zero8f() { v8f z = {}; return z; }

// --- CDNA5 async global->LDS copy (ASYNCcnt path, 16B granules) ---
__device__ inline void async_copy_b128(void* lds_dst, const void* gbl_src) {
    unsigned lds = (unsigned)(size_t)lds_dst;   // low 32 bits = LDS byte address
    asm volatile("global_load_async_to_lds_b128 %0, %1, off"
                 :: "v"(lds), "v"(gbl_src) : "memory");
}
__device__ inline void async_wait0() {
    asm volatile("s_wait_asynccnt 0x0" ::: "memory");
}

__device__ inline void atomicMaxF(float* addr, float v) {
    if (v >= 0.0f) atomicMax((int*)addr, __float_as_int(v));
    else           atomicMin((unsigned int*)addr, __float_as_uint(v));
}

// ---------------------------------------------------------------------------
// Transpose-convert f32 weight W[fi][fo] -> f16 Wt[fo][kPad] (zero-padded K)
// ---------------------------------------------------------------------------
__global__ void w_to_f16t(const float* __restrict__ W, _Float16* __restrict__ Wt,
                          int fi, int fo, int kPad) {
    int idx = blockIdx.x * blockDim.x + threadIdx.x;
    if (idx >= fo * kPad) return;
    int c = idx / kPad, k = idx % kPad;
    Wt[idx] = (_Float16)(k < fi ? W[k * fo + c] : 0.0f);
}

// ---------------------------------------------------------------------------
// Generic WMMA GEMM: C[M][Nc] = act(A[M][K] * Bt[Nc][K]^T + bias)
// block (32,8): 8 waves; each wave computes 32 rows x 64 cols (2 strips x 4
// wmma tiles) so every staged-LDS B fragment feeds two WMMAs.
// B tile (64 cols x K) staged once per block in LDS via async loads.
// A row index clamped to M-1 -> unconditional global b128 loads in hot loop.
// act: 0 none, 1 relu, 2 tanh.  Dynamic LDS = 64*K*2 bytes.
// ---------------------------------------------------------------------------
__global__ __launch_bounds__(256)
void wmma_gemm(const _Float16* __restrict__ A, const _Float16* __restrict__ Bt,
               const float* __restrict__ bias, _Float16* __restrict__ outH,
               float* __restrict__ outF, int M, int K, int Nc, int act) {
    extern __shared__ char smemG[];
    _Float16* sBt = (_Float16*)smemG;            // [64][K]

    const int lane = threadIdx.x;
    const int wid  = threadIdx.y;
    const int nb0  = blockIdx.x * 64;
    const int kch  = K >> 3;                     // 16B chunks per row

    // ---- stage B tile: rows by wave, chunks by lane (async -> LDS) ----
    for (int rr = wid; rr < 64; rr += 8) {
        int c = nb0 + rr;
        if (c < Nc) {
            for (int ck = lane; ck < kch; ck += 32)
                async_copy_b128(sBt + rr * K + ck * 8,
                                Bt + (size_t)c * K + ck * 8);
        } else {
            for (int ck = lane; ck < kch; ck += 32)
                *(v8h*)(sBt + rr * K + ck * 8) = zero8h();
        }
    }
    async_wait0();
    __syncthreads();

    const int half = lane >> 4;
    const int n16  = lane & 15;
    const int row0 = (blockIdx.y * 8 + wid) * 32;
    if (row0 >= M) return;                       // safe: no barriers below
    // clamped A rows -> unconditional loads (dup rows for OOB, never stored)
    const int ar0 = min(row0 + n16,      M - 1);
    const int ar1 = min(row0 + 16 + n16, M - 1);
    const _Float16* ap0 = A + (size_t)ar0 * K + 8 * half;
    const _Float16* ap1 = A + (size_t)ar1 * K + 8 * half;

    v8f acc[2][4];
    #pragma unroll
    for (int s = 0; s < 2; s++)
        #pragma unroll
        for (int t = 0; t < 4; t++) acc[s][t] = zero8f();

    for (int k0 = 0; k0 < K; k0 += 32) {
        Frag16 a0, a1;
        a0.h.lo = *(const v8h*)(ap0 + k0);
        a0.h.hi = *(const v8h*)(ap0 + k0 + 16);
        a1.h.lo = *(const v8h*)(ap1 + k0);
        a1.h.hi = *(const v8h*)(ap1 + k0 + 16);
        #pragma unroll
        for (int t = 0; t < 4; t++) {
            const _Float16* bp = sBt + (16 * t + n16) * K + k0;
            Frag16 b;
            b.h.lo = *(const v8h*)(bp + 8 * half);
            b.h.hi = *(const v8h*)(bp + 16 + 8 * half);
            acc[0][t] = __builtin_amdgcn_wmma_f32_16x16x32_f16(
                false, a0.v, false, b.v, (short)0, acc[0][t], false, false);
            acc[1][t] = __builtin_amdgcn_wmma_f32_16x16x32_f16(
                false, a1.v, false, b.v, (short)0, acc[1][t], false, false);
        }
    }

    #pragma unroll
    for (int s = 0; s < 2; s++) {
        #pragma unroll
        for (int t = 0; t < 4; t++) {
            int c = nb0 + 16 * t + n16;
            if (c >= Nc) continue;
            float bv = bias ? bias[c] : 0.0f;
            #pragma unroll
            for (int r = 0; r < 8; r++) {
                int row = row0 + 16 * s + r + 8 * half;
                if (row >= M) continue;
                float v = acc[s][t][r] + bv;
                if (act == 1) v = fmaxf(v, 0.0f);
                else if (act == 2) v = tanhf(v);
                size_t o = (size_t)row * Nc + c;
                if (outH) outH[o] = (_Float16)v;
                if (outF) outF[o] = v;
            }
        }
    }
}

// ---------------------------------------------------------------------------
// Fused ECC: edge features -> 4-layer MLP (WMMA, LDS-staged weights) ->
// einsum with x[send] -> atomic segment_sum into hacc[recv][64].
// block (32,8): 8 waves x 16 edges. Weights staged via async global->LDS.
// ---------------------------------------------------------------------------
#define ECC_W16_HALFS 34816   // W1t 64x32 | W2t 64x64 | W3t 64x64 | W4t 384x64
#define ECC_SMEM_BYTES (ECC_W16_HALFS*2 + 576*4 + 8*16*64*2 + 8*6*16*4 + 8*16*4)

__global__ __launch_bounds__(256)
void ecc_fused(const float* __restrict__ x, const int* __restrict__ ei,
               const _Float16* __restrict__ w16,
               const float* __restrict__ b1, const float* __restrict__ b2,
               const float* __restrict__ b3, const float* __restrict__ b4,
               float* __restrict__ hacc) {
    extern __shared__ char smem[];
    _Float16* sW     = (_Float16*)smem;                       // 34816 halfs
    float*    sB     = (float*)(sW + ECC_W16_HALFS);          // 576 floats
    _Float16* sStage = (_Float16*)(sB + 576);                 // 8*16*64 halfs
    float*    sXs    = (float*)(sStage + 8 * 16 * 64);        // 8*6*16
    int*      sRv    = (int*)(sXs + 8 * 6 * 16);              // 8*16

    const int tid = threadIdx.y * 32 + threadIdx.x;
    // ---- async weight stage: 69632B = 4352 x b128 ----
    for (int i = tid; i < ECC_W16_HALFS / 8; i += 256)
        async_copy_b128(sW + i * 8, w16 + i * 8);
    for (int i = tid; i < 576; i += 256) {
        float v;
        if (i < 64)       v = b1[i];
        else if (i < 128) v = b2[i - 64];
        else if (i < 192) v = b3[i - 128];
        else              v = b4[i - 192];
        sB[i] = v;
    }
    async_wait0();
    __syncthreads();

    const _Float16* sW1 = sW;
    const _Float16* sW2 = sW + 2048;
    const _Float16* sW3 = sW + 6144;
    const _Float16* sW4 = sW + 10240;

    const int lane = threadIdx.x, wid = threadIdx.y;
    const int half = lane >> 4, n16 = lane & 15;
    const int et = (blockIdx.x * 8 + wid) * 16;
    _Float16* stg = sStage + wid * 16 * 64;
    float* xs = sXs + wid * 96;
    int* rvp = sRv + wid * 16;

    // ---- edge features (lanes 0..15, one edge each) ----
    if (lane < 16) {
        int e = et + lane;
        float feat[7] = {0, 0, 0, 0, 0, 0, 0};
        int rv = 0;
        if (e < NEDGES) {
            int s = ei[2 * e], r = ei[2 * e + 1];
            rv = r;
            float dx[6];
            #pragma unroll
            for (int f = 0; f < 6; f++)
                dx[f] = x[(size_t)r * 6 + f] - x[(size_t)s * 6 + f];
            float d = sqrtf(dx[0]*dx[0] + dx[1]*dx[1] + dx[2]*dx[2]);
            float inv = d > 0.0f ? 1.0f / d : 0.0f;
            feat[0] = dx[3]; feat[1] = dx[4]; feat[2] = dx[5]; feat[3] = d;
            feat[4] = dx[0]*inv; feat[5] = dx[1]*inv; feat[6] = dx[2]*inv;
            #pragma unroll
            for (int f = 0; f < 6; f++) xs[f * 16 + lane] = x[(size_t)s * 6 + f];
        } else {
            #pragma unroll
            for (int f = 0; f < 6; f++) xs[f * 16 + lane] = 0.0f;
        }
        rvp[lane] = rv;
        #pragma unroll
        for (int k = 0; k < 7; k++)  stg[lane * 64 + k] = (_Float16)feat[k];
        #pragma unroll
        for (int k = 7; k < 32; k++) stg[lane * 64 + k] = (_Float16)0.0f;
    }
    __syncthreads();

    v8f acc[4];

    // ---- layer 1: 16x32 @ 32x64 ----
    {
        Frag16 a;
        a.h.lo = *(const v8h*)(stg + n16 * 64 + 8 * half);
        a.h.hi = *(const v8h*)(stg + n16 * 64 + 16 + 8 * half);
        #pragma unroll
        for (int t = 0; t < 4; t++) {
            Frag16 b;
            const _Float16* bp = sW1 + (16 * t + n16) * 32;
            b.h.lo = *(const v8h*)(bp + 8 * half);
            b.h.hi = *(const v8h*)(bp + 16 + 8 * half);
            acc[t] = __builtin_amdgcn_wmma_f32_16x16x32_f16(
                false, a.v, false, b.v, (short)0, zero8f(), false, false);
        }
    }
    __syncthreads();
    #pragma unroll
    for (int t = 0; t < 4; t++) {
        float bv = sB[16 * t + n16];
        #pragma unroll
        for (int r = 0; r < 8; r++)
            stg[(r + 8 * half) * 64 + 16 * t + n16] =
                (_Float16)fmaxf(acc[t][r] + bv, 0.0f);
    }
    __syncthreads();

    // ---- layers 2,3: 16x64 @ 64x64 ----
    #pragma unroll
    for (int L = 0; L < 2; L++) {
        const _Float16* Wt = (L == 0) ? sW2 : sW3;
        const float* bb = sB + 64 + L * 64;
        Frag16 a0, a1;
        a0.h.lo = *(const v8h*)(stg + n16 * 64 + 8 * half);
        a0.h.hi = *(const v8h*)(stg + n16 * 64 + 16 + 8 * half);
        a1.h.lo = *(const v8h*)(stg + n16 * 64 + 32 + 8 * half);
        a1.h.hi = *(const v8h*)(stg + n16 * 64 + 48 + 8 * half);
        #pragma unroll
        for (int t = 0; t < 4; t++) {
            const _Float16* bp = Wt + (16 * t + n16) * 64;
            Frag16 b0, b1f;
            b0.h.lo  = *(const v8h*)(bp + 8 * half);
            b0.h.hi  = *(const v8h*)(bp + 16 + 8 * half);
            b1f.h.lo = *(const v8h*)(bp + 32 + 8 * half);
            b1f.h.hi = *(const v8h*)(bp + 48 + 8 * half);
            acc[t] = __builtin_amdgcn_wmma_f32_16x16x32_f16(
                false, a0.v, false, b0.v, (short)0, zero8f(), false, false);
            acc[t] = __builtin_amdgcn_wmma_f32_16x16x32_f16(
                false, a1.v, false, b1f.v, (short)0, acc[t], false, false);
        }
        __syncthreads();
        #pragma unroll
        for (int t = 0; t < 4; t++) {
            float bv = bb[16 * t + n16];
            #pragma unroll
            for (int r = 0; r < 8; r++)
                stg[(r + 8 * half) * 64 + 16 * t + n16] =
                    (_Float16)fmaxf(acc[t][r] + bv, 0.0f);
        }
        __syncthreads();
    }

    // ---- layer 4 (64 -> 384) fused with einsum over f=0..5 ----
    Frag16 a0, a1;
    a0.h.lo = *(const v8h*)(stg + n16 * 64 + 8 * half);
    a0.h.hi = *(const v8h*)(stg + n16 * 64 + 16 + 8 * half);
    a1.h.lo = *(const v8h*)(stg + n16 * 64 + 32 + 8 * half);
    a1.h.hi = *(const v8h*)(stg + n16 * 64 + 48 + 8 * half);

    v8f macc[4];
    #pragma unroll
    for (int t = 0; t < 4; t++) macc[t] = zero8f();

    for (int f = 0; f < 6; f++) {
        float sR[8];
        #pragma unroll
        for (int r = 0; r < 8; r++) sR[r] = xs[f * 16 + r + 8 * half];
        #pragma unroll
        for (int t = 0; t < 4; t++) {
            int cc = f * 64 + 16 * t + n16;
            const _Float16* bp = sW4 + (size_t)cc * 64;
            Frag16 b0, b1f;
            b0.h.lo  = *(const v8h*)(bp + 8 * half);
            b0.h.hi  = *(const v8h*)(bp + 16 + 8 * half);
            b1f.h.lo = *(const v8h*)(bp + 32 + 8 * half);
            b1f.h.hi = *(const v8h*)(bp + 48 + 8 * half);
            v8f k = __builtin_amdgcn_wmma_f32_16x16x32_f16(
                false, a0.v, false, b0.v, (short)0, zero8f(), false, false);
            k = __builtin_amdgcn_wmma_f32_16x16x32_f16(
                false, a1.v, false, b1f.v, (short)0, k, false, false);
            float bv = sB[192 + cc];
            #pragma unroll
            for (int r = 0; r < 8; r++) macc[t][r] += sR[r] * (k[r] + bv);
        }
    }

    // ---- scatter: hacc[recv][c] += msgs ----
    #pragma unroll
    for (int r = 0; r < 8; r++) {
        int m = r + 8 * half;
        if (et + m < NEDGES) {
            float* dst = hacc + (size_t)rvp[m] * 64;
            #pragma unroll
            for (int t = 0; t < 4; t++)
                atomicAdd(dst + 16 * t + n16, macc[t][r]);
        }
    }
}

// ---------------------------------------------------------------------------
__global__ void ecc_root_update(const float* __restrict__ hacc,
                                const float* __restrict__ x,
                                const float* __restrict__ root,
                                const float* __restrict__ bias,
                                _Float16* __restrict__ h) {
    int idx = blockIdx.x * blockDim.x + threadIdx.x;
    if (idx >= NNODES * 64) return;
    int node = idx >> 6, c = idx & 63;
    float s = hacc[idx] + bias[c];
    #pragma unroll
    for (int f = 0; f < 6; f++) s += x[node * 6 + f] * root[f * 64 + c];
    h[idx] = (_Float16)fmaxf(s, 0.0f);
}

// acc[send[e]][c] += a_vals[e] * hw[recv[e]][c]   (f16 gather, f32 atomics)
__global__ void edge_scatter(const _Float16* __restrict__ hw,
                             const int* __restrict__ ei,
                             const float* __restrict__ av,
                             float* __restrict__ acc, int fo, int logChunks) {
    int tid = blockIdx.x * blockDim.x + threadIdx.x;
    int e = tid >> logChunks;
    if (e >= NEDGES) return;
    int c = (tid & ((1 << logChunks) - 1)) * 8;
    int s = ei[2 * e], r = ei[2 * e + 1];
    float a = av[e];
    v8h v = *(const v8h*)(hw + (size_t)r * fo + c);
    float* dst = acc + (size_t)s * fo + c;
    #pragma unroll
    for (int j = 0; j < 8; j++) atomicAdd(dst + j, a * (float)v[j]);
}

__global__ void bias_relu_h(const float* __restrict__ acc,
                            const float* __restrict__ b,
                            _Float16* __restrict__ out, int n, int foMask) {
    int idx = blockIdx.x * blockDim.x + threadIdx.x;
    if (idx >= n) return;
    out[idx] = (_Float16)fmaxf(acc[idx] + b[idx & foMask], 0.0f);
}

__global__ void fill_f32(float* p, float v, int n) {
    int idx = blockIdx.x * blockDim.x + threadIdx.x;
    if (idx < n) p[idx] = v;
}

__global__ void seg_max(const _Float16* __restrict__ h,
                        const int* __restrict__ gi, float* __restrict__ g) {
    int idx = blockIdx.x * blockDim.x + threadIdx.x;
    if (idx >= NNODES * 512) return;
    int node = idx >> 9, c = idx & 511;
    atomicMaxF(g + (size_t)gi[node] * 512 + c, (float)h[idx]);
}

__global__ void cvt_f16(const float* __restrict__ s, _Float16* __restrict__ d, int n) {
    int idx = blockIdx.x * blockDim.x + threadIdx.x;
    if (idx < n) d[idx] = (_Float16)s[idx];
}

// ---------------------------------------------------------------------------
extern "C" void kernel_launch(void* const* d_in, const int* in_sizes, int n_in,
                              void* d_out, int out_size, void* d_ws, size_t ws_size,
                              hipStream_t stream) {
    (void)in_sizes; (void)n_in; (void)out_size; (void)ws_size;
    const float* x    = (const float*)d_in[0];
    const int*   ei   = (const int*)d_in[1];
    const float* av   = (const float*)d_in[2];
    const int*   gi   = (const int*)d_in[3];
    const float* eW[4] = {(const float*)d_in[4], (const float*)d_in[5],
                          (const float*)d_in[6], (const float*)d_in[7]};
    const float* eB[4] = {(const float*)d_in[8], (const float*)d_in[9],
                          (const float*)d_in[10], (const float*)d_in[11]};
    const float* root  = (const float*)d_in[12];
    const float* rbias = (const float*)d_in[13];
    const float* gW[4] = {(const float*)d_in[14], (const float*)d_in[15],
                          (const float*)d_in[16], (const float*)d_in[17]};
    const float* gB[4] = {(const float*)d_in[18], (const float*)d_in[19],
                          (const float*)d_in[20], (const float*)d_in[21]};
    const float* dW[6] = {(const float*)d_in[22], (const float*)d_in[23],
                          (const float*)d_in[24], (const float*)d_in[25],
                          (const float*)d_in[26], (const float*)d_in[27]};
    const float* dB[6] = {(const float*)d_in[28], (const float*)d_in[29],
                          (const float*)d_in[30], (const float*)d_in[31],
                          (const float*)d_in[32], (const float*)d_in[33]};
    const float* d2w = (const float*)d_in[34];
    const float* d2b = (const float*)d_in[35];
    float* out = (float*)d_out;

    char* ws = (char*)d_ws;
    size_t off = 0;
    auto alloc = [&](size_t bytes) -> char* {
        char* p = ws + off;
        off += (bytes + 255) & ~(size_t)255;
        return p;
    };

    _Float16* eccW16 = (_Float16*)alloc((size_t)ECC_W16_HALFS * 2);
    _Float16* gWt[4]; int gfi[4] = {64, 64, 128, 256}, gfo[4] = {64, 128, 256, 512};
    for (int l = 0; l < 4; l++) gWt[l] = (_Float16*)alloc((size_t)gfo[l] * gfi[l] * 2);
    _Float16* dWt[6]; int dk[6] = {512, 512, 256, 128, 64, 64}, dn[6] = {512, 256, 128, 64, 64, 64};
    for (int l = 0; l < 6; l++) dWt[l] = (_Float16*)alloc((size_t)dn[l] * dk[l] * 2);
    _Float16* d2t  = (_Float16*)alloc(6 * 64 * 2);
    float*    hacc = (float*)alloc((size_t)NNODES * 512 * 4);
    _Float16* hA   = (_Float16*)alloc((size_t)NNODES * 512 * 2);
    _Float16* hB   = (_Float16*)alloc((size_t)NNODES * 512 * 2);
    float*    g    = (float*)alloc((size_t)NGRAPH * 512 * 4);
    _Float16* gh0  = (_Float16*)alloc((size_t)NGRAPH * 512 * 2);
    _Float16* gh1  = (_Float16*)alloc((size_t)NGRAPH * 512 * 2);

    const int T = 256;
    auto cvt = [&](const float* W, _Float16* Wt, int fi, int fo, int kp) {
        int n = fo * kp;
        w_to_f16t<<<(n + T - 1) / T, T, 0, stream>>>(W, Wt, fi, fo, kp);
    };

    // --- weight conversion (transposed f16) ---
    cvt(eW[0], eccW16 + 0,     7, 64, 32);
    cvt(eW[1], eccW16 + 2048, 64, 64, 64);
    cvt(eW[2], eccW16 + 6144, 64, 64, 64);
    cvt(eW[3], eccW16 + 10240, 64, 384, 64);
    for (int l = 0; l < 4; l++) cvt(gW[l], gWt[l], gfi[l], gfo[l], gfi[l]);
    for (int l = 0; l < 6; l++) cvt(dW[l], dWt[l], dk[l], dn[l], dk[l]);
    cvt(d2w, d2t, 64, 6, 64);

    dim3 blk(32, 8);

    // --- ECC conv ---
    hipMemsetAsync(hacc, 0, (size_t)NNODES * 64 * 4, stream);
    ecc_fused<<<dim3((NEDGES + 127) / 128), blk, ECC_SMEM_BYTES, stream>>>(
        x, ei, eccW16, eB[0], eB[1], eB[2], eB[3], hacc);
    ecc_root_update<<<(NNODES * 64 + T - 1) / T, T, 0, stream>>>(hacc, x, root, rbias, hA);

    // --- GCN layers ---
    for (int l = 0; l < 4; l++) {
        int fi = gfi[l], fo = gfo[l];
        dim3 grid((fo + 63) / 64, (NNODES + 255) / 256);
        wmma_gemm<<<grid, blk, (size_t)64 * fi * 2, stream>>>(
            hA, gWt[l], nullptr, hB, nullptr, NNODES, fi, fo, 0);
        hipMemsetAsync(hacc, 0, (size_t)NNODES * fo * 4, stream);
        int logChunks = (fo == 64) ? 3 : (fo == 128) ? 4 : (fo == 256) ? 5 : 6;
        int tot = NEDGES << logChunks;
        edge_scatter<<<(tot + T - 1) / T, T, 0, stream>>>(hB, ei, av, hacc, fo, logChunks);
        bias_relu_h<<<(NNODES * fo + T - 1) / T, T, 0, stream>>>(hacc, gB[l], hA,
                                                                 NNODES * fo, fo - 1);
    }

    // --- segment max ---
    fill_f32<<<(NGRAPH * 512 + T - 1) / T, T, 0, stream>>>(g, -3.0e38f, NGRAPH * 512);
    seg_max<<<(NNODES * 512 + T - 1) / T, T, 0, stream>>>(hA, gi, g);
    cvt_f16<<<(NGRAPH * 512 + T - 1) / T, T, 0, stream>>>(g, gh0, NGRAPH * 512);

    // --- decoder (tanh layers, ping-pong) ---
    _Float16* src = gh0; _Float16* dst = gh1;
    for (int l = 0; l < 6; l++) {
        dim3 grid((dn[l] + 63) / 64, (NGRAPH + 255) / 256);
        wmma_gemm<<<grid, blk, (size_t)64 * dk[l] * 2, stream>>>(
            src, dWt[l], dB[l], dst, nullptr, NGRAPH, dk[l], dn[l], 2);
        _Float16* tmp = src; src = dst; dst = tmp;
    }
    // --- final linear 64 -> 6 (f32 out) ---
    wmma_gemm<<<dim3(1, (NGRAPH + 255) / 256), blk, (size_t)64 * 64 * 2, stream>>>(
        src, d2t, d2b, nullptr, out, NGRAPH, 64, 6, 0);
}
